// FactorPredictor_44418551775295
// MI455X (gfx1250) — matrix operand order repaired and checked by reading, not compiled, not run on hardware
//
#include <hip/hip_runtime.h>
#include <math.h>

#define H      512
#define NPTS   4096
#define NHEAD  64
#define LEAKY  0.1f
#define EPSN   1e-12f
#define EPSS   1e-9f
#define WPAD   8

typedef __attribute__((ext_vector_type(16))) _Float16 v16h;
typedef __attribute__((ext_vector_type(8)))  _Float16 v8h;
typedef __attribute__((ext_vector_type(4)))  _Float16 v4h;
typedef __attribute__((ext_vector_type(8)))  float    v8f;
typedef __attribute__((ext_vector_type(4)))  float    v4f;

#if defined(__has_builtin)
#  if __has_builtin(__builtin_amdgcn_s_wait_asynccnt)
#    define WAIT_ASYNC(n) __builtin_amdgcn_s_wait_asynccnt(n)
#  endif
#endif
#ifndef WAIT_ASYNC
#  define WAIT_ASYNC(n) asm volatile("s_wait_asynccnt %0" :: "i"(n) : "memory")
#endif

// async 16B copy global -> LDS (CDNA5, ASYNCcnt-tracked, no VGPR data round-trip)
__device__ __forceinline__ void async_copy16(const _Float16* __restrict__ gsrc,
                                             _Float16* ldst) {
    unsigned int ldsoff = (unsigned int)(unsigned long long)ldst; // low 32b = LDS offset
    unsigned long long ga = (unsigned long long)gsrc;
    asm volatile("global_load_async_to_lds_b128 %0, %1, off"
                 :: "v"(ldsoff), "v"(ga) : "memory");
}

// ---------------- kernel P: fp32 -> f16 pre-conversion ----------------
__global__ __launch_bounds__(256) void cvt_f16_kernel(const float* __restrict__ src,
                                                      _Float16* __restrict__ dst,
                                                      int n4) {
    int i = blockIdx.x * blockDim.x + threadIdx.x;
    if (i < n4) {
        v4f f = *(const v4f*)(src + (size_t)i * 4);
        v4h h;
        h[0] = (_Float16)f[0]; h[1] = (_Float16)f[1];
        h[2] = (_Float16)f[2]; h[3] = (_Float16)f[3];
        *(v4h*)(dst + (size_t)i * 4) = h;
    }
}

// ---------------- kernel 0: qn = q / max(||q||, eps) ----------------
__global__ __launch_bounds__(256) void qnorm_kernel(const float* __restrict__ q,
                                                    float* __restrict__ qn) {
    const int head = blockIdx.x;
    const int t = threadIdx.x;
    __shared__ float red[256];
    float a0 = q[head * H + t];
    float a1 = q[head * H + t + 256];
    red[t] = a0 * a0 + a1 * a1;
    __syncthreads();
    for (int s = 128; s > 0; s >>= 1) {
        if (t < s) red[t] += red[t + s];
        __syncthreads();
    }
    float inv = 1.0f / fmaxf(sqrtf(red[0]), EPSN);
    qn[head * H + t]       = a0 * inv;
    qn[head * H + t + 256] = a1 * inv;
}

// ---------------- kernel 1: WMMA scores pass ----------------
// k[n,d] = sum_h e[n,h]*Wk[head,d,h]; score[n] = max(qn.k_n,0)/max(||k_n||,eps)
// grid: (N/128, NHEAD), block: 256 (8 waves, 16 rows each)
__global__ __launch_bounds__(256) void scores_kernel(const _Float16* __restrict__ e16,
                                                     const _Float16* __restrict__ wk16,
                                                     const float* __restrict__ qn,
                                                     float* __restrict__ scores) {
    const int head   = blockIdx.y;
    const int wave   = threadIdx.x >> 5;
    const int lane   = threadIdx.x & 31;
    const int laneLo = lane & 15;
    const int hiHalf = (lane >= 16) ? 1 : 0;
    const int rowBase = blockIdx.x * 128 + wave * 16;
    const int myRow   = rowBase + laneLo;

    // double-buffered 16(d) x 512(h) Wk tile, f16, padded rows (bank spread)
    __shared__ _Float16 sW[2][16 * (H + WPAD)];
    __shared__ float sQ[H];                 // this head's qn, staged once

    sQ[threadIdx.x]       = qn[head * H + threadIdx.x];
    sQ[threadIdx.x + 256] = qn[head * H + threadIdx.x + 256];

    const _Float16* wkHead = wk16 + (size_t)head * H * H;

    // issue async staging of tile `dt` into buffer `buf` (4 x 16B per thread)
    auto stage = [&](int dt, int buf) {
        const int dbase = dt * 16;
        #pragma unroll
        for (int j = 0; j < 4; ++j) {
            const int g   = j * 256 + threadIdx.x;   // 1024 granules of 16B
            const int row = g >> 6;
            const int c16 = g & 63;
            async_copy16(wkHead + (size_t)(dbase + row) * H + c16 * 8,
                         &sW[buf][row * (H + WPAD) + c16 * 8]);
        }
    };

    // ---- resident A fragments: this wave's 16 e-rows (f16, loaded once) ----
    v16h A[16];
    const _Float16* eRow = e16 + (size_t)myRow * H;
    #pragma unroll
    for (int s = 0; s < 16; ++s) {
        const int h0 = s * 32 + (hiHalf ? 8 : 0);
        v8h a0 = *(const v8h*)(eRow + h0);
        v8h a1 = *(const v8h*)(eRow + h0 + 16);
        A[s] = __builtin_shufflevector(a0, a1, 0, 1, 2, 3, 4, 5, 6, 7,
                                               8, 9, 10, 11, 12, 13, 14, 15);
    }

    float psq[8], pdt[8];
    #pragma unroll
    for (int i = 0; i < 8; ++i) { psq[i] = 0.0f; pdt[i] = 0.0f; }

    stage(0, 0);   // prologue

    for (int dt = 0; dt < 32; ++dt) {
        const int buf = dt & 1;
        if (dt + 1 < 32) {
            stage(dt + 1, buf ^ 1);      // overlap next tile's copy with this compute
            WAIT_ASYNC(4);               // oldest 4 ops (tile dt) complete, in order
        } else {
            WAIT_ASYNC(0);
        }
        __syncthreads();                 // tile dt visible to all waves

        const _Float16* bbase = &sW[buf][laneLo * (H + WPAD) + (hiHalf ? 8 : 0)];
        auto loadB = [&](int s) -> v16h {
            const _Float16* bp = bbase + s * 32;
            v8h b0 = *(const v8h*)bp;
            v8h b1 = *(const v8h*)(bp + 16);
            return __builtin_shufflevector(b0, b1, 0, 1, 2, 3, 4, 5, 6, 7,
                                                   8, 9, 10, 11, 12, 13, 14, 15);
        };

        // 16 rows(n) x 16 cols(d) tile over h=512; B pipelined two steps deep
        const float qd = sQ[dt * 16 + laneLo];   // col d is lane-constant (ds load)
        v8f c = {};
        v16h b0v = loadB(0);
        v16h b1v = loadB(1);
        #pragma unroll
        for (int s = 0; s < 16; ++s) {
            if (s & 1) {
                c = __builtin_amdgcn_wmma_f32_16x16x32_f16(false, A[s], false, b1v,
                                                           (short)0, c, false, false);
                if (s + 2 < 16) b1v = loadB(s + 2);
            } else {
                c = __builtin_amdgcn_wmma_f32_16x16x32_f16(false, A[s], false, b0v,
                                                           (short)0, c, false, false);
                if (s + 2 < 16) b0v = loadB(s + 2);
            }
        }

        #pragma unroll
        for (int i = 0; i < 8; ++i) {
            psq[i] += c[i] * c[i];
            pdt[i] += c[i] * qd;
        }
        __syncthreads();                 // all reads of sW[buf] done before refill
    }

    // reduce across the 16 lanes of each half (rows i / i+8 per C-layout)
    #pragma unroll
    for (int i = 0; i < 8; ++i) {
        float sq = psq[i], dp = pdt[i];
        for (int m = 8; m >= 1; m >>= 1) {
            sq += __shfl_xor(sq, m, 32);
            dp += __shfl_xor(dp, m, 32);
        }
        psq[i] = sq; pdt[i] = dp;
    }
    if (laneLo == 0) {
        const int row0 = rowBase + (hiHalf ? 8 : 0);
        #pragma unroll
        for (int i = 0; i < 8; ++i) {
            float nrm = fmaxf(sqrtf(psq[i]), EPSN);
            scores[(size_t)head * NPTS + row0 + i] = fmaxf(pdt[i], 0.0f) / nrm;
        }
    }
}

// ---------------- kernel 2: weights + ew[k,:] = weights[k] . e ----------------
__global__ __launch_bounds__(256) void weights_ew_kernel(const float* __restrict__ e,
                                                         const float* __restrict__ scores,
                                                         float* __restrict__ ew) {
    const int head = blockIdx.x;
    const int t = threadIdx.x;
    __shared__ float sw[NPTS];
    __shared__ float red[256];

    float local = 0.0f;
    for (int n = t; n < NPTS; n += 256) {
        float v = scores[(size_t)head * NPTS + n];
        sw[n] = v;
        local += v;
    }
    red[t] = local;
    __syncthreads();
    for (int s = 128; s > 0; s >>= 1) {
        if (t < s) red[t] += red[t + s];
        __syncthreads();
    }
    const float ssum = red[0];
    const float invs = 1.0f / fmaxf(ssum, EPSS);
    const bool uni = ssum < EPSS;
    for (int n = t; n < NPTS; n += 256)
        sw[n] = uni ? (1.0f / NPTS) : sw[n] * invs;
    __syncthreads();

    float acc0 = 0.0f, acc1 = 0.0f;
    for (int n = 0; n < NPTS; ++n) {
        const float w = sw[n];
        acc0 += w * e[(size_t)n * H + t];
        acc1 += w * e[(size_t)n * H + t + 256];
    }
    ew[head * H + t]       = acc0;
    ew[head * H + t + 256] = acc1;
}

// ---------------- kernel 3: h_muti = Wv[k].ew ; MLP ; mu, sigma ----------------
__global__ __launch_bounds__(256) void head_mlp_kernel(const float* __restrict__ Wv,
                                                       const float* __restrict__ ew,
                                                       const float* __restrict__ Wh,
                                                       const float* __restrict__ bh,
                                                       const float* __restrict__ Wmu,
                                                       const float* __restrict__ bmu,
                                                       const float* __restrict__ Wsig,
                                                       const float* __restrict__ bsig,
                                                       float* __restrict__ out) {
    const int head = blockIdx.x;
    const int t = threadIdx.x;
    const int wave = t >> 5;
    const int lane = t & 31;
    __shared__ float sX[H];
    __shared__ float sM[H];
    __shared__ float sHH[H];

    sX[t]       = ew[head * H + t];
    sX[t + 256] = ew[head * H + t + 256];
    __syncthreads();

    const float* WvH = Wv + (size_t)head * H * H;
    for (int d = wave; d < H; d += 8) {
        const float* row = WvH + (size_t)d * H;
        float acc = 0.0f;
        for (int h = lane; h < H; h += 32) acc += row[h] * sX[h];
        for (int m = 16; m >= 1; m >>= 1) acc += __shfl_xor(acc, m, 32);
        if (lane == 0) sM[d] = acc;
    }
    __syncthreads();

    for (int d = wave; d < H; d += 8) {
        const float* row = Wh + (size_t)d * H;
        float acc = 0.0f;
        for (int h = lane; h < H; h += 32) acc += row[h] * sM[h];
        for (int m = 16; m >= 1; m >>= 1) acc += __shfl_xor(acc, m, 32);
        if (lane == 0) {
            float x = acc + bh[d];
            sHH[d] = (x > 0.0f) ? x : LEAKY * x;
        }
    }
    __syncthreads();

    if (wave == 0) {
        float am = 0.0f, as = 0.0f;
        for (int h = lane; h < H; h += 32) {
            am += Wmu[h] * sHH[h];
            as += Wsig[h] * sHH[h];
        }
        for (int m = 16; m >= 1; m >>= 1) {
            am += __shfl_xor(am, m, 32);
            as += __shfl_xor(as, m, 32);
        }
        if (lane == 0) {
            float mu = am + bmu[0];
            float x  = as + bsig[0];
            float sp = (x > 20.0f) ? x : log1pf(expf(x));
            out[head]         = mu;
            out[NHEAD + head] = sp;
        }
    }
}

extern "C" void kernel_launch(void* const* d_in, const int* in_sizes, int n_in,
                              void* d_out, int out_size, void* d_ws, size_t ws_size,
                              hipStream_t stream) {
    const float* e    = (const float*)d_in[0];
    const float* Wk   = (const float*)d_in[1];
    const float* Wv   = (const float*)d_in[2];
    const float* q    = (const float*)d_in[3];
    const float* Wh   = (const float*)d_in[4];
    const float* bh   = (const float*)d_in[5];
    const float* Wmu  = (const float*)d_in[6];
    const float* bmu  = (const float*)d_in[7];
    const float* Wsig = (const float*)d_in[8];
    const float* bsig = (const float*)d_in[9];
    float* out = (float*)d_out;

    // workspace layout: floats first, then f16 regions (16B aligned)
    float* wsf    = (float*)d_ws;
    float* qn     = wsf;                              // 64*512
    float* scores = qn + NHEAD * H;                   // 64*4096
    float* ew     = scores + (size_t)NHEAD * NPTS;    // 64*512
    _Float16* e16  = (_Float16*)(ew + NHEAD * H);     // 4096*512
    _Float16* wk16 = e16 + (size_t)NPTS * H;          // 64*512*512

    const int n4e = NPTS * H / 4;             // 524288
    const int n4w = NHEAD * H * H / 4;        // 4194304
    cvt_f16_kernel<<<n4e / 256, 256, 0, stream>>>(e, e16, n4e);
    cvt_f16_kernel<<<n4w / 256, 256, 0, stream>>>(Wk, wk16, n4w);

    qnorm_kernel<<<NHEAD, 256, 0, stream>>>(q, qn);
    scores_kernel<<<dim3(NPTS / 128, NHEAD), 256, 0, stream>>>(e16, wk16, qn, scores);
    weights_ew_kernel<<<NHEAD, 256, 0, stream>>>(e, scores, ew);
    head_mlp_kernel<<<NHEAD, 256, 0, stream>>>(Wv, ew, Wh, bh, Wmu, bmu, Wsig, bsig, out);
}
